// GRUNetMultiLabel_50766513439266
// MI455X (gfx1250) — matrix operand order repaired and checked by reading, not compiled, not run on hardware
//
#include <hip/hip_runtime.h>
#include <hip/hip_bf16.h>

// Problem dims (fixed by the reference)
#define BB 64
#define TT 256
#define VV 50000
#define DD 512
#define HH 1024
#define OO 64
#define GG (3 * HH)          // 3072
#define ROWS (BB * TT)       // 16384

#define NCHUNK 8             // hidden-column chunks of 128 for the recurrence
#define GRU_NWG (4 * NCHUNK) // 32 persistent workgroups

typedef __attribute__((ext_vector_type(16))) __bf16 v16bf;
typedef __attribute__((ext_vector_type(8)))  __bf16 v8bf;
typedef __attribute__((ext_vector_type(8)))  float  v8f;

// -------- helpers --------
__device__ __forceinline__ __bf16 f2bf(float f) {
  union { float f; unsigned u; } v; v.f = f;
  unsigned r = v.u + 0x7FFFu + ((v.u >> 16) & 1u);   // round-to-nearest-even
  union { unsigned short s; __bf16 b; } o;
  o.s = (unsigned short)(r >> 16);
  return o.b;
}

__device__ __forceinline__ float sigmoidf_(float v) {
  return 1.0f / (1.0f + __expf(-v));
}

__device__ __forceinline__ v8f wmma_bf16(v16bf a, v16bf b, v8f c) {
  // (neg_a, A, neg_b, B, c_mod, C, reuse_a, reuse_b)
  return __builtin_amdgcn_wmma_f32_16x16x32_bf16(false, a, false, b, (short)0, c,
                                                 false, false);
}

// A fragment (16x32 bf16, M x K): lane m = lane&15, K-halves by lane>>4.
// rowp points at element [m][0] of a row contiguous in K.
__device__ __forceinline__ v16bf load_a_row(const __bf16* rowp, int k0, int kb) {
  union { v16bf v; v8bf h[2]; } u;
  u.h[0] = *(const v8bf*)(rowp + k0 + kb);        // K = k0+kb   .. +7
  u.h[1] = *(const v8bf*)(rowp + k0 + 16 + kb);   // K = k0+16+kb.. +7
  return u.v;
}

// B fragment (32x16 bf16, K x N): lane L holds col n = n0+(L&15), K = k0+kb..+7, k0+16+kb..+7 (kb = (L>>4)*8).
// Wt is K-major: Wt[k][n], leading dim ldw.
__device__ __forceinline__ v16bf load_b_tile(const __bf16* Wt, int ldw,
                                             int k0, int n0, int lane) {
  const __bf16* p = Wt + n0 + (lane & 15);
  const int kb = (lane >> 4) * 8;
  v16bf v;
#pragma unroll
  for (int s = 0; s < 8; ++s) {
    v[s] = p[(size_t)(k0 + kb + s) * ldw];
    v[8 + s] = p[(size_t)(k0 + 16 + kb + s) * ldw];
  }
  return v;
}

// Device-wide monotonic arrive/spin barrier (persistent grid, all WGs resident).
__device__ __forceinline__ void gbar(unsigned* cnt, unsigned target, int tid) {
  __threadfence();               // release: my h stores visible at device scope
  __syncthreads();               // whole WG arrived
  if (tid == 0) {
    __hip_atomic_fetch_add(cnt, 1u, __ATOMIC_RELEASE, __HIP_MEMORY_SCOPE_AGENT);
    while (__hip_atomic_load(cnt, __ATOMIC_ACQUIRE, __HIP_MEMORY_SCOPE_AGENT) <
           target) {
      __builtin_amdgcn_s_sleep(2);
    }
  }
  __syncthreads();
  __threadfence();               // acquire: see other WGs' h stores
}

// -------- K0: convert + transpose weights to bf16, K-major --------
__global__ void prep_weights(const float* __restrict__ W_ih,
                             const float* __restrict__ W_hh,
                             const float* __restrict__ W_fc,
                             __bf16* __restrict__ wt_ih,   // [D][G]
                             __bf16* __restrict__ wt_hh,   // [H][G]
                             __bf16* __restrict__ wt_fc) { // [H][O]
  const int NIH = DD * GG, NHH = HH * GG, NFC = HH * OO;
  const int total = NIH + NHH + NFC;
  for (int i = blockIdx.x * blockDim.x + threadIdx.x; i < total;
       i += gridDim.x * blockDim.x) {
    if (i < NIH) {
      int k = i / GG, g = i % GG;
      wt_ih[i] = f2bf(W_ih[(size_t)g * DD + k]);
    } else if (i < NIH + NHH) {
      int j = i - NIH;
      int k = j / GG, g = j % GG;
      wt_hh[j] = f2bf(W_hh[(size_t)g * HH + k]);
    } else {
      int j = i - NIH - NHH;
      int k = j / OO, o = j % OO;
      wt_fc[j] = f2bf(W_fc[(size_t)o * HH + k]);
    }
  }
}

// -------- K1: embedding gather + input GEMM: gates_x = xe @ W_ih^T + b_ih ----
// grid: (ROWS/16, 4 N-chunks of 768), block 256 (8 waves, 6 tiles per wave)
#define LDA 520   // padded bf16 row stride (16B-aligned, bank-spread)
__global__ __launch_bounds__(256) void gemm_input(
    const int* __restrict__ x, const float* __restrict__ emb,
    const __bf16* __restrict__ wt_ih, const float* __restrict__ b_ih,
    float* __restrict__ gates_x) {
  __shared__ __bf16 xe[16][LDA];
  const int mt = blockIdx.x;
  const int chunk = blockIdx.y;
  const int tid = threadIdx.x, lane = tid & 31, w = tid >> 5;

  // gather 16 token embeddings, convert f32 -> bf16 into LDS
  for (int idx = tid; idx < 16 * DD; idx += 256) {
    int m = idx >> 9, k = idx & (DD - 1);
    int row = mt * 16 + m;
    int tok = x[row];
    xe[m][k] = f2bf(emb[(size_t)tok * DD + k]);
  }
  __syncthreads();

  const int mlane = lane & 15, kb = (lane >> 4) * 8;
  v8f vz = {};
  v8f acc[6];
#pragma unroll
  for (int i = 0; i < 6; ++i) acc[i] = vz;

  for (int k0 = 0; k0 < DD; k0 += 32) {
    v16bf a = load_a_row(&xe[mlane][0], k0, kb);
    v16bf bt6[6];                 // distinct buffers -> one load clause
#pragma unroll
    for (int i = 0; i < 6; ++i) {
      int n0 = chunk * 768 + (w * 6 + i) * 16;
      bt6[i] = load_b_tile(wt_ih, GG, k0, n0, lane);
    }
#pragma unroll
    for (int i = 0; i < 6; ++i) acc[i] = wmma_bf16(a, bt6[i], acc[i]);
  }

  const int rbase = mt * 16 + (lane >> 4) * 8;
#pragma unroll
  for (int i = 0; i < 6; ++i) {
    int n = chunk * 768 + (w * 6 + i) * 16 + mlane;
    float bv = b_ih[n];
#pragma unroll
    for (int r = 0; r < 8; ++r)
      gates_x[(size_t)(rbase + r) * GG + n] = acc[i][r] + bv;
  }
}

// -------- K2: persistent GRU recurrence over 32 WGPs -----------------------
// grid: (4 batch-tiles, 8 column-chunks of 128), block 256 (8 waves).
// Wave w owns hidden cols [nc*128 + w*16, +16): accs for r/z/n tiles.
// h ping-pongs between two global bf16 buffers (L2-hot); one device barrier
// per step orders "write h(t)" before "read h(t)".
__global__ __launch_bounds__(256) void gru_seq(
    const float* __restrict__ gates_x, const __bf16* __restrict__ wt_hh,
    const float* __restrict__ b_hh, __bf16* __restrict__ hs,
    __bf16* __restrict__ hbuf,          // 2 x [64][1024]
    unsigned* __restrict__ sync_cnt) {  // zeroed by hipMemsetAsync
  const int bt = blockIdx.x;            // 0..3
  const int nc = blockIdx.y;            // 0..7
  const int b0 = bt * 16;
  const int tid = threadIdx.x, lane = tid & 31, w = tid >> 5;
  const int mlane = lane & 15, kb = (lane >> 4) * 8, mrow = (lane >> 4) * 8;
  const int ncol = nc * 128 + w * 16;   // this wave's hidden column base

  __bf16* h0 = hbuf;
  __bf16* h1 = hbuf + (size_t)BB * HH;

  // h(-1) = 0: each WG zeroes its own (bt, nc) chunk of h0
  for (int idx = tid; idx < 16 * 128; idx += 256) {
    int m = idx >> 7, c = idx & 127;
    h0[(size_t)(b0 + m) * HH + nc * 128 + c] = f2bf(0.0f);
  }
  gbar(sync_cnt, GRU_NWG, tid);         // episode 1

  float hreg[8] = {};
  v8f vz = {};
  for (int t = 0; t < TT; ++t) {
    const __bf16* hin = (t & 1) ? h1 : h0;
    __bf16* hout      = (t & 1) ? h0 : h1;
    const __bf16* arow = hin + (size_t)(b0 + mlane) * HH;

    v8f accr = vz, accz = vz, accn = vz;
    // software pipeline: stage k0+32 while computing k0
    v16bf a  = load_a_row(arow, 0, kb);
    v16bf br = load_b_tile(wt_hh, GG, 0, 0 * HH + ncol, lane);
    v16bf bz = load_b_tile(wt_hh, GG, 0, 1 * HH + ncol, lane);
    v16bf bn = load_b_tile(wt_hh, GG, 0, 2 * HH + ncol, lane);
#pragma unroll 2
    for (int k0 = 0; k0 < HH; k0 += 32) {
      const int kn = (k0 + 32 < HH) ? k0 + 32 : 0;   // branch-free rotate
      v16bf a2  = load_a_row(arow, kn, kb);
      v16bf br2 = load_b_tile(wt_hh, GG, kn, 0 * HH + ncol, lane);
      v16bf bz2 = load_b_tile(wt_hh, GG, kn, 1 * HH + ncol, lane);
      v16bf bn2 = load_b_tile(wt_hh, GG, kn, 2 * HH + ncol, lane);
      accr = wmma_bf16(a, br, accr);
      accz = wmma_bf16(a, bz, accz);
      accn = wmma_bf16(a, bn, accn);
      a = a2; br = br2; bz = bz2; bn = bn2;
    }

    // gate math in the C-fragment layout (lane holds col n, rows mrow..+7)
    {
      int n = ncol + mlane;
      float bhr = b_hh[n], bhz = b_hh[HH + n], bhn = b_hh[2 * HH + n];
#pragma unroll
      for (int r = 0; r < 8; ++r) {
        int bi = b0 + mrow + r;
        size_t grow = ((size_t)bi * TT + t) * GG;
        float xr = gates_x[grow + n];
        float xz = gates_x[grow + HH + n];
        float xn = gates_x[grow + 2 * HH + n];
        float rg = sigmoidf_(xr + accr[r] + bhr);
        float zg = sigmoidf_(xz + accz[r] + bhz);
        float ng = tanhf(xn + rg * (accn[r] + bhn));
        float h = (1.0f - zg) * ng + zg * hreg[r];
        hreg[r] = h;
        __bf16 hb = f2bf(h);
        hout[(size_t)bi * HH + n] = hb;
        hs[((size_t)bi * TT + t) * HH + n] = hb;
      }
    }
    if (t + 1 < TT) gbar(sync_cnt, (unsigned)(t + 2) * GRU_NWG, tid);
  }
}

// -------- K3: output GEMM + sigmoid + threshold ----------------------------
// grid: ROWS/16, block 128 (4 waves, one 16-col N tile each; O = 64)
__global__ __launch_bounds__(128) void gemm_out(
    const __bf16* __restrict__ hs, const __bf16* __restrict__ wt_fc,
    const float* __restrict__ b_fc, float* __restrict__ out) {
  const int mt = blockIdx.x;
  const int tid = threadIdx.x, lane = tid & 31, w = tid >> 5;
  const int mlane = lane & 15, kb = (lane >> 4) * 8;
  const int mrow = (lane >> 4) * 8;

  const __bf16* arow = hs + (size_t)(mt * 16 + mlane) * HH;
  v8f acc = {};
  v16bf a = load_a_row(arow, 0, kb);
  v16bf b = load_b_tile(wt_fc, OO, 0, w * 16, lane);
#pragma unroll 2
  for (int k0 = 0; k0 < HH; k0 += 32) {
    const int kn = (k0 + 32 < HH) ? k0 + 32 : 0;
    v16bf a2 = load_a_row(arow, kn, kb);
    v16bf b2 = load_b_tile(wt_fc, OO, kn, w * 16, lane);
    acc = wmma_bf16(a, b, acc);
    a = a2; b = b2;
  }

  const int n = w * 16 + mlane;
  const float bv = b_fc[n];
  const size_t OFF = (size_t)ROWS * OO;
#pragma unroll
  for (int r = 0; r < 8; ++r) {
    int row = mt * 16 + mrow + r;
    float p = sigmoidf_(acc[r] + bv);
    out[(size_t)row * OO + n] = p;
    out[OFF + (size_t)row * OO + n] = (p > 0.5f) ? 1.0f : 0.0f;
  }
}

// -------- launch ------------------------------------------------------------
extern "C" void kernel_launch(void* const* d_in, const int* in_sizes, int n_in,
                              void* d_out, int out_size, void* d_ws, size_t ws_size,
                              hipStream_t stream) {
  const int*   x    = (const int*)d_in[0];
  const float* emb  = (const float*)d_in[1];
  const float* W_ih = (const float*)d_in[2];
  const float* W_hh = (const float*)d_in[3];
  const float* b_ih = (const float*)d_in[4];
  const float* b_hh = (const float*)d_in[5];
  const float* W_fc = (const float*)d_in[6];
  const float* b_fc = (const float*)d_in[7];
  float* out = (float*)d_out;

  // workspace layout (bytes)
  const size_t WT_IH_B = (size_t)DD * GG * 2;        //  3,145,728
  const size_t WT_HH_B = (size_t)HH * GG * 2;        //  6,291,456
  const size_t WT_FC_B = (size_t)HH * OO * 2;        //    131,072
  const size_t GATES_OFF = ((WT_IH_B + WT_HH_B + WT_FC_B + 255) / 256) * 256;
  const size_t GATES_B = (size_t)ROWS * GG * 4;      // 201,326,592
  const size_t HS_OFF  = GATES_OFF + GATES_B;
  const size_t HS_B    = (size_t)ROWS * HH * 2;      //  33,554,432
  const size_t HB_OFF  = HS_OFF + HS_B;
  const size_t HB_B    = (size_t)2 * BB * HH * 2;    //     262,144
  const size_t CNT_OFF = HB_OFF + HB_B;

  char* ws = (char*)d_ws;
  __bf16*   wt_ih = (__bf16*)(ws);
  __bf16*   wt_hh = (__bf16*)(ws + WT_IH_B);
  __bf16*   wt_fc = (__bf16*)(ws + WT_IH_B + WT_HH_B);
  float*    gates = (float*)(ws + GATES_OFF);
  __bf16*   hsbuf = (__bf16*)(ws + HS_OFF);
  __bf16*   hbuf  = (__bf16*)(ws + HB_OFF);
  unsigned* cnt   = (unsigned*)(ws + CNT_OFF);

  prep_weights<<<2048, 256, 0, stream>>>(W_ih, W_hh, W_fc, wt_ih, wt_hh, wt_fc);
  gemm_input<<<dim3(ROWS / 16, 4), 256, 0, stream>>>(x, emb, wt_ih, b_ih, gates);
  hipMemsetAsync(cnt, 0, sizeof(unsigned), stream);  // barrier counter = 0
  gru_seq<<<dim3(4, NCHUNK), 256, 0, stream>>>(gates, wt_hh, b_hh, hsbuf, hbuf, cnt);
  gemm_out<<<ROWS / 16, 128, 0, stream>>>(hsbuf, wt_fc, b_fc, out);
}
